// ComplexNetMLP_86569360818367
// MI455X (gfx1250) — 8-bit matrix operand order repaired and checked by reading, compiled, not run on hardware
//
#include <hip/hip_runtime.h>
#include <hip/hip_bf16.h>


#define HID   2048
#define IMD   5504
#define MTOK  8192
#define KSTEP 64

typedef __attribute__((ext_vector_type(8))) int          v8i;
typedef __attribute__((ext_vector_type(2))) unsigned int u32x2;
typedef __attribute__((ext_vector_type(4))) unsigned int u32x4;
typedef __attribute__((ext_vector_type(4))) unsigned int tdm_g0;
typedef __attribute__((ext_vector_type(4))) int          tdm_g2;

#if __has_builtin(__builtin_amdgcn_tensor_load_to_lds) && \
    __has_builtin(__builtin_amdgcn_s_wait_tensorcnt)
#define HAVE_TDM 1
#else
#define HAVE_TDM 0
#endif

__device__ __forceinline__ v8i wmma_iu8(v8i a, v8i b, v8i c) {
  return __builtin_amdgcn_wmma_i32_16x16x64_iu8(true, a, true, b, c, false, false);
}

#if HAVE_TDM
__device__ __forceinline__ void tdm_tile_2d(const signed char* g,
                                            const signed char* lds,
                                            int rows, int stride) {
  unsigned long long ga = (unsigned long long)(size_t)g;
  tdm_g0 g0;
  g0[0] = 1u;
  g0[1] = (unsigned)(size_t)lds;
  g0[2] = (unsigned)ga;
  g0[3] = (unsigned)((ga >> 32) & 0x01ffffffu)
          | (2u << 30);
  v8i g1;
  g1[0] = 0;
  g1[1] = (int)(64u << 16);
  g1[2] = (int)(((unsigned)rows & 0xffffu) << 16);
  g1[3] = (int)(64u << 16);
  g1[4] = (int)((unsigned)rows & 0xffffu);
  g1[5] = stride;
  g1[6] = 0;
  g1[7] = 0;
  tdm_g2 z4 = (tdm_g2)0;
  v8i    z8 = (v8i)0;
  __builtin_amdgcn_tensor_load_to_lds(g0, g1, z4, z4, z8, 0);
}
#endif

__device__ __forceinline__ v8i lds_a_frag(const signed char* row, int hi) {
  const u32x2* q = (const u32x2*)row;
  u32x2 a = q[2 * hi], b = q[2 * hi + 1], c = q[4 + 2 * hi], d = q[5 + 2 * hi];
  v8i f = { (int)a.x, (int)a.y, (int)b.x, (int)b.y,
            (int)c.x, (int)c.y, (int)d.x, (int)d.y };
  return f;
}

__device__ __forceinline__ v8i lds_b_frag(const signed char* row, int hi) {
  const u32x4* q = (const u32x4*)row;
  u32x4 lo = q[hi], hh = q[2 + hi];
  v8i f = { (int)lo.x, (int)lo.y, (int)lo.z, (int)lo.w,
            (int)hh.x, (int)hh.y, (int)hh.z, (int)hh.w };
  return f;
}

__device__ __forceinline__ float block_reduce_sum(float v, float* sh) {
  int t = threadIdx.x;
  sh[t] = v; __syncthreads();
  for (int s = 128; s > 0; s >>= 1) { if (t < s) sh[t] += sh[t + s]; __syncthreads(); }
  float r = sh[0]; __syncthreads();
  return r;
}
__device__ __forceinline__ float block_reduce_max(float v, float* sh) {
  int t = threadIdx.x;
  sh[t] = v; __syncthreads();
  for (int s = 128; s > 0; s >>= 1) { if (t < s) sh[t] = fmaxf(sh[t], sh[t + s]); __syncthreads(); }
  float r = sh[0]; __syncthreads();
  return r;
}

__device__ __forceinline__ signed char q8(float x) {
  float r = rintf(x);
  r = fminf(fmaxf(r, -128.f), 127.f);
  return (signed char)(int)r;
}

__global__ __launch_bounds__(256) void k_wquant_sums(
    const float* __restrict__ wr0, const float* __restrict__ wi0,
    const float* __restrict__ wr1, const float* __restrict__ wi1,
    const float* __restrict__ wr2, const float* __restrict__ wi2,
    float* __restrict__ sums) {
  __shared__ float sh[256];
  const int mat = blockIdx.y;
  const float* wr = (mat == 0) ? wr0 : (mat == 1) ? wr1 : wr2;
  const float* wi = (mat == 0) ? wi0 : (mat == 1) ? wi1 : wi2;
  const size_t N = (size_t)IMD * HID;
  float ra = 0.f, rc = 0.f, ia = 0.f, ic = 0.f;
  for (size_t idx = (size_t)blockIdx.x * 256 + threadIdx.x; idx < N;
       idx += (size_t)gridDim.x * 256) {
    float ar = fabsf(wr[idx]), ai = fabsf(wi[idx]);
    if (ar >= ai) { ra += ar; rc += 1.f; }
    else          { ia += ai; ic += 1.f; }
  }
  ra = block_reduce_sum(ra, sh);
  rc = block_reduce_sum(rc, sh);
  ia = block_reduce_sum(ia, sh);
  ic = block_reduce_sum(ic, sh);
  if (threadIdx.x == 0) {
    atomicAdd(&sums[mat * 4 + 0], ra);
    atomicAdd(&sums[mat * 4 + 1], rc);
    atomicAdd(&sums[mat * 4 + 2], ia);
    atomicAdd(&sums[mat * 4 + 3], ic);
  }
}

__global__ void k_means(const float* __restrict__ sums, float* __restrict__ means) {
  int w = threadIdx.x;
  if (w < 3) {
    means[w * 2 + 0] = fmaxf(sums[w * 4 + 0] / fmaxf(sums[w * 4 + 1], 1.f), 1e-5f);
    means[w * 2 + 1] = fmaxf(sums[w * 4 + 2] / fmaxf(sums[w * 4 + 3], 1.f), 1e-5f);
  }
}

__global__ __launch_bounds__(256) void k_ternarize(
    const float* __restrict__ wr0, const float* __restrict__ wi0,
    const float* __restrict__ wr1, const float* __restrict__ wi1,
    const float* __restrict__ wr2, const float* __restrict__ wi2,
    signed char* __restrict__ qr0, signed char* __restrict__ qi0,
    signed char* __restrict__ qr1, signed char* __restrict__ qi1,
    signed char* __restrict__ qr2, signed char* __restrict__ qi2) {
  const int mat = blockIdx.y;
  const float* wr = (mat == 0) ? wr0 : (mat == 1) ? wr1 : wr2;
  const float* wi = (mat == 0) ? wi0 : (mat == 1) ? wi1 : wi2;
  signed char* qr = (mat == 0) ? qr0 : (mat == 1) ? qr1 : qr2;
  signed char* qi = (mat == 0) ? qi0 : (mat == 1) ? qi1 : qi2;
  const size_t N = (size_t)IMD * HID;
  for (size_t idx = (size_t)blockIdx.x * 256 + threadIdx.x; idx < N;
       idx += (size_t)gridDim.x * 256) {
    float r = wr[idx], i = wi[idx];
    signed char tr = 0, ti = 0;
    if (fabsf(r) >= fabsf(i)) tr = (r >= 0.f) ? 1 : -1;
    else                      ti = (i >= 0.f) ? 1 : -1;
    qr[idx] = tr; qi[idx] = ti;
  }
}

__global__ __launch_bounds__(256) void k_actquant_x(
    const float* __restrict__ xr, const float* __restrict__ xi,
    signed char* __restrict__ qr, signed char* __restrict__ qi,
    float* __restrict__ rs_arr, float* __restrict__ is_arr) {
  __shared__ float sh[256];
  const int m = blockIdx.x, t = threadIdx.x;
  const float* xrp = xr + (size_t)m * HID;
  const float* xip = xi + (size_t)m * HID;
  float mr = 0.f, mi = 0.f;
  for (int j = t; j < HID; j += 256) {
    mr = fmaxf(mr, fabsf(xrp[j]));
    mi = fmaxf(mi, fabsf(xip[j]));
  }
  mr = block_reduce_max(mr, sh);
  mi = block_reduce_max(mi, sh);
  const float rs  = 127.f / fmaxf(mr, 1e-5f);
  const float is_ = 127.f / fmaxf(mi, 1e-5f);
  for (int j = t; j < HID; j += 256) {
    qr[(size_t)m * HID + j] = q8(xrp[j] * rs);
    qi[(size_t)m * HID + j] = q8(xip[j] * is_);
  }
  if (t == 0) { rs_arr[m] = rs; is_arr[m] = is_; }
}

__global__ __launch_bounds__(256) void k_gateup(
    const signed char* __restrict__ qxr, const signed char* __restrict__ qxi,
    const signed char* __restrict__ wgr, const signed char* __restrict__ wgi,
    const signed char* __restrict__ wur, const signed char* __restrict__ wui,
    const float* __restrict__ rs_arr, const float* __restrict__ is_arr,
    const float* __restrict__ means,
    float* __restrict__ h_r, float* __restrict__ h_i) {
  __shared__ __align__(128) signed char lA[2][2][32][64];
  __shared__ __align__(128) signed char lW[2][4][64][64];
  const int t = threadIdx.x;
  const int m0 = blockIdx.y * 32;
  const int n0 = blockIdx.x * 64;
  const int lane = t & 31, wave = t >> 5;
  const int wm = (wave >> 2) << 4, wn = (wave & 3) << 4;
  const int lm = lane & 15, hi = lane >> 4;

  const signed char* wp[4] = { wgr, wgi, wur, wui };
  const signed char* ap[2] = { qxr, qxi };

  v8i aGrr = (v8i)0, aGii = (v8i)0, aGri = (v8i)0, aGir = (v8i)0;
  v8i aUrr = (v8i)0, aUii = (v8i)0, aUri = (v8i)0, aUir = (v8i)0;

  constexpr int KIT = HID / KSTEP;

#if HAVE_TDM
  if (t < 32) {
#pragma unroll
    for (int s = 0; s < 2; ++s)
      tdm_tile_2d(ap[s] + (size_t)m0 * HID, &lA[0][s][0][0], 32, HID);
#pragma unroll
    for (int mat = 0; mat < 4; ++mat)
      tdm_tile_2d(wp[mat] + (size_t)n0 * HID, &lW[0][mat][0][0], 64, HID);
  }
#else
  {
    const int amat = t >> 7, aidx = t & 127;
    const int arow = aidx >> 2, acol = (aidx & 3) << 4;
    *(u32x4*)&lA[0][amat][arow][acol] =
        *(const u32x4*)(ap[amat] + (size_t)(m0 + arow) * HID + acol);
    const int wrow = t >> 2, wcol = (t & 3) << 4;
#pragma unroll
    for (int mat = 0; mat < 4; ++mat)
      *(u32x4*)&lW[0][mat][wrow][wcol] =
          *(const u32x4*)(wp[mat] + (size_t)(n0 + wrow) * HID + wcol);
  }
#endif

  for (int k = 0; k < KIT; ++k) {
    const int cur = k & 1;
    const int k1 = (k + 1) * KSTEP;
#if HAVE_TDM
    if (t < 32) {
      if (k + 1 < KIT) {
#pragma unroll
        for (int s = 0; s < 2; ++s)
          tdm_tile_2d(ap[s] + (size_t)m0 * HID + k1, &lA[cur ^ 1][s][0][0], 32, HID);
#pragma unroll
        for (int mat = 0; mat < 4; ++mat)
          tdm_tile_2d(wp[mat] + (size_t)n0 * HID + k1, &lW[cur ^ 1][mat][0][0], 64, HID);
        __builtin_amdgcn_s_wait_tensorcnt(6);
      } else {
        __builtin_amdgcn_s_wait_tensorcnt(0);
      }
    }
#else
    if (k + 1 < KIT) {
      const int amat = t >> 7, aidx = t & 127;
      const int arow = aidx >> 2, acol = (aidx & 3) << 4;
      *(u32x4*)&lA[cur ^ 1][amat][arow][acol] =
          *(const u32x4*)(ap[amat] + (size_t)(m0 + arow) * HID + k1 + acol);
      const int wrow = t >> 2, wcol = (t & 3) << 4;
#pragma unroll
      for (int mat = 0; mat < 4; ++mat)
        *(u32x4*)&lW[cur ^ 1][mat][wrow][wcol] =
            *(const u32x4*)(wp[mat] + (size_t)(n0 + wrow) * HID + k1 + wcol);
    }
#endif
    __syncthreads();
    v8i ar  = lds_a_frag(&lA[cur][0][wm + lm][0], hi);
    v8i ai  = lds_a_frag(&lA[cur][1][wm + lm][0], hi);
    v8i bgr = lds_b_frag(&lW[cur][0][wn + lm][0], hi);
    v8i bgi = lds_b_frag(&lW[cur][1][wn + lm][0], hi);
    v8i bur = lds_b_frag(&lW[cur][2][wn + lm][0], hi);
    v8i bui = lds_b_frag(&lW[cur][3][wn + lm][0], hi);
    aGrr = wmma_iu8(ar, bgr, aGrr);
    aGii = wmma_iu8(ai, bgi, aGii);
    aGri = wmma_iu8(ar, bgi, aGri);
    aGir = wmma_iu8(ai, bgr, aGir);
    aUrr = wmma_iu8(ar, bur, aUrr);
    aUii = wmma_iu8(ai, bui, aUii);
    aUri = wmma_iu8(ar, bui, aUri);
    aUir = wmma_iu8(ai, bur, aUir);
    __syncthreads();
  }

  const float gmr = means[0], gmi = means[1], umr = means[2], umi = means[3];
  const int n = n0 + wn + lm;
#pragma unroll
  for (int v = 0; v < 8; ++v) {
    const int m = m0 + wm + (hi << 3) + v;
    const float rs = rs_arr[m], is_ = is_arr[m];
    const float gr = (float)aGrr[v] * (gmr / rs) + (float)aGii[v] * (gmi / is_);
    const float gi = (float)aGri[v] * (gmi / rs) - (float)aGir[v] * (gmr / is_);
    const float ur = (float)aUrr[v] * (umr / rs) + (float)aUii[v] * (umi / is_);
    const float ui = (float)aUri[v] * (umi / rs) - (float)aUir[v] * (umr / is_);
    const bool msk = (gr < 0.f) && (gi < 0.f);
    float arr = msk ? 0.f : gr;
    float aii = msk ? 0.f : gi;
    arr *= arr; aii *= aii;
    h_r[(size_t)m * IMD + n] = arr * ur + aii * ui;
    h_i[(size_t)m * IMD + n] = arr * ui - aii * ur;
  }
}

__global__ __launch_bounds__(256) void k_norm_quant(
    const float* __restrict__ h_r, const float* __restrict__ h_i,
    const float* __restrict__ gwr, const float* __restrict__ gwi,
    signed char* __restrict__ qnr, signed char* __restrict__ qni,
    float* __restrict__ rsn_arr, float* __restrict__ isn_arr) {
  __shared__ float sh[256];
  const int m = blockIdx.x, t = threadIdx.x;
  const float* hr = h_r + (size_t)m * IMD;
  const float* hi = h_i + (size_t)m * IMD;
  float ss = 0.f, mr = 0.f, mi = 0.f;
  for (int j = t; j < IMD; j += 256) {
    float r = hr[j], i = hi[j];
    ss += r * r + i * i;
    mr = fmaxf(mr, fabsf(gwr[j] * r));
    mi = fmaxf(mi, fabsf(gwi[j] * i));
    if (j + 256 < IMD) { __builtin_prefetch(&hr[j + 256], 0, 0);
                         __builtin_prefetch(&hi[j + 256], 0, 0); }
  }
  ss = block_reduce_sum(ss, sh);
  mr = block_reduce_max(mr, sh);
  mi = block_reduce_max(mi, sh);
  const float inv = rsqrtf(ss / (float)IMD + 1e-6f);
  const float rsn = 127.f / fmaxf(mr * inv, 1e-5f);
  const float isn = 127.f / fmaxf(mi * inv, 1e-5f);
  for (int j = t; j < IMD; j += 256) {
    qnr[(size_t)m * IMD + j] = q8(gwr[j] * hr[j] * inv * rsn);
    qni[(size_t)m * IMD + j] = q8(gwi[j] * hi[j] * inv * isn);
  }
  if (t == 0) { rsn_arr[m] = rsn; isn_arr[m] = isn; }
}

__global__ __launch_bounds__(256) void k_down(
    const signed char* __restrict__ qnr, const signed char* __restrict__ qni,
    const signed char* __restrict__ wdr, const signed char* __restrict__ wdi,
    const float* __restrict__ rs_arr, const float* __restrict__ is_arr,
    const float* __restrict__ means,
    float* __restrict__ out) {
  __shared__ __align__(128) signed char lA[2][2][32][64];
  __shared__ __align__(128) signed char lW[2][2][64][64];
  const int t = threadIdx.x;
  const int m0 = blockIdx.y * 32;
  const int n0 = blockIdx.x * 64;
  const int lane = t & 31, wave = t >> 5;
  const int wm = (wave >> 2) << 4, wn = (wave & 3) << 4;
  const int lm = lane & 15, hi = lane >> 4;

  const signed char* wp[2] = { wdr, wdi };
  const signed char* ap[2] = { qnr, qni };

  v8i aRR = (v8i)0, aII = (v8i)0, aRI = (v8i)0, aIR = (v8i)0;

  constexpr int KIT = IMD / KSTEP;

#if HAVE_TDM
  if (t < 32) {
#pragma unroll
    for (int s = 0; s < 2; ++s) {
      tdm_tile_2d(ap[s] + (size_t)m0 * IMD, &lA[0][s][0][0], 32, IMD);
      tdm_tile_2d(wp[s] + (size_t)n0 * IMD, &lW[0][s][0][0], 64, IMD);
    }
  }
#else
  {
    const int amat = t >> 7, aidx = t & 127;
    const int arow = aidx >> 2, acol = (aidx & 3) << 4;
    *(u32x4*)&lA[0][amat][arow][acol] =
        *(const u32x4*)(ap[amat] + (size_t)(m0 + arow) * IMD + acol);
    const int wrow = t >> 2, wcol = (t & 3) << 4;
#pragma unroll
    for (int mat = 0; mat < 2; ++mat)
      *(u32x4*)&lW[0][mat][wrow][wcol] =
          *(const u32x4*)(wp[mat] + (size_t)(n0 + wrow) * IMD + wcol);
  }
#endif

  for (int k = 0; k < KIT; ++k) {
    const int cur = k & 1;
    const int k1 = (k + 1) * KSTEP;
#if HAVE_TDM
    if (t < 32) {
      if (k + 1 < KIT) {
#pragma unroll
        for (int s = 0; s < 2; ++s) {
          tdm_tile_2d(ap[s] + (size_t)m0 * IMD + k1, &lA[cur ^ 1][s][0][0], 32, IMD);
          tdm_tile_2d(wp[s] + (size_t)n0 * IMD + k1, &lW[cur ^ 1][s][0][0], 64, IMD);
        }
        __builtin_amdgcn_s_wait_tensorcnt(4);
      } else {
        __builtin_amdgcn_s_wait_tensorcnt(0);
      }
    }
#else
    if (k + 1 < KIT) {
      const int amat = t >> 7, aidx = t & 127;
      const int arow = aidx >> 2, acol = (aidx & 3) << 4;
      *(u32x4*)&lA[cur ^ 1][amat][arow][acol] =
          *(const u32x4*)(ap[amat] + (size_t)(m0 + arow) * IMD + k1 + acol);
      const int wrow = t >> 2, wcol = (t & 3) << 4;
#pragma unroll
      for (int mat = 0; mat < 2; ++mat)
        *(u32x4*)&lW[cur ^ 1][mat][wrow][wcol] =
            *(const u32x4*)(wp[mat] + (size_t)(n0 + wrow) * IMD + k1 + wcol);
    }
#endif
    __syncthreads();
    v8i ar  = lds_a_frag(&lA[cur][0][wm + lm][0], hi);
    v8i ai  = lds_a_frag(&lA[cur][1][wm + lm][0], hi);
    v8i bdr = lds_b_frag(&lW[cur][0][wn + lm][0], hi);
    v8i bdi = lds_b_frag(&lW[cur][1][wn + lm][0], hi);
    aRR = wmma_iu8(ar, bdr, aRR);
    aII = wmma_iu8(ai, bdi, aII);
    aRI = wmma_iu8(ar, bdi, aRI);
    aIR = wmma_iu8(ai, bdr, aIR);
    __syncthreads();
  }

  const float dmr = means[4], dmi = means[5];
  const int n = n0 + wn + lm;
#pragma unroll
  for (int v = 0; v < 8; ++v) {
    const int m = m0 + wm + (hi << 3) + v;
    const float rs = rs_arr[m], is_ = is_arr[m];
    out[(size_t)m * HID + n] =
        (float)aRR[v] * (dmr / rs) + (float)aII[v] * (dmi / is_);
    out[(size_t)MTOK * HID + (size_t)m * HID + n] =
        (float)aRI[v] * (dmi / rs) - (float)aIR[v] * (dmr / is_);
  }
}

extern "C" void kernel_launch(void* const* d_in, const int* in_sizes, int n_in,
                              void* d_out, int out_size, void* d_ws, size_t ws_size,
                              hipStream_t stream) {
  const float* x_r  = (const float*)d_in[0];
  const float* x_i  = (const float*)d_in[1];
  const float* g_wr = (const float*)d_in[2];
  const float* g_wi = (const float*)d_in[3];
  const float* u_wr = (const float*)d_in[4];
  const float* u_wi = (const float*)d_in[5];
  const float* dw_r = (const float*)d_in[6];
  const float* dw_i = (const float*)d_in[7];
  const float* ln_r = (const float*)d_in[8];
  const float* ln_i = (const float*)d_in[9];
  float* out = (float*)d_out;

  char* ws = (char*)d_ws;
  constexpr size_t WB = (size_t)IMD * HID;
  constexpr size_t XB = (size_t)MTOK * HID;
  constexpr size_t HB = (size_t)MTOK * IMD;
  size_t off = 0;
  float* sums  = (float*)(ws + off); off += 256;
  float* means = (float*)(ws + off); off += 256;
  signed char* qw[6];
  for (int i = 0; i < 6; ++i) { qw[i] = (signed char*)(ws + off); off += WB; }
  signed char* qxr = (signed char*)(ws + off); off += XB;
  signed char* qxi = (signed char*)(ws + off); off += XB;
  float* rs_arr  = (float*)(ws + off); off += (size_t)MTOK * 4;
  float* is_arr  = (float*)(ws + off); off += (size_t)MTOK * 4;
  float* h_r     = (float*)(ws + off); off += HB * 4;
  float* h_i     = (float*)(ws + off); off += HB * 4;
  signed char* qnr = (signed char*)(ws + off); off += HB;
  signed char* qni = (signed char*)(ws + off); off += HB;
  float* rsn_arr = (float*)(ws + off); off += (size_t)MTOK * 4;
  float* isn_arr = (float*)(ws + off); off += (size_t)MTOK * 4;

  (void)hipMemsetAsync(sums, 0, 64, stream);

  dim3 blk(256);
  k_wquant_sums<<<dim3(512, 3), blk, 0, stream>>>(g_wr, g_wi, u_wr, u_wi,
                                                  dw_r, dw_i, sums);
  k_means<<<1, 32, 0, stream>>>(sums, means);
  k_ternarize<<<dim3(1024, 3), blk, 0, stream>>>(g_wr, g_wi, u_wr, u_wi, dw_r, dw_i,
                                                 qw[0], qw[1], qw[2], qw[3], qw[4], qw[5]);
  k_actquant_x<<<MTOK, blk, 0, stream>>>(x_r, x_i, qxr, qxi, rs_arr, is_arr);
  k_gateup<<<dim3(IMD / 64, MTOK / 32), blk, 0, stream>>>(
      qxr, qxi, qw[0], qw[1], qw[2], qw[3], rs_arr, is_arr, means, h_r, h_i);
  k_norm_quant<<<MTOK, blk, 0, stream>>>(h_r, h_i, ln_r, ln_i, qnr, qni,
                                         rsn_arr, isn_arr);
  k_down<<<dim3(HID / 64, MTOK / 32), blk, 0, stream>>>(
      qnr, qni, qw[4], qw[5], rsn_arr, isn_arr, means, out);
}
